// Node_GCN_24275155157237
// MI455X (gfx1250) — matrix operand order repaired and checked by reading, not compiled, not run on hardware
//
#include <hip/hip_runtime.h>

// ---------------------------------------------------------------------------
// Node GCN on MI455X (gfx1250, wave32, WMMA).
// out[n,j,h] = f(x)[n,j,h] + sum_i edge[n,i,j] * g(cat(x,x))[n,i,h]
// GEMMs via v_wmma_f32_16x16x32_bf16; edge streamed with a 4-deep
// global_load_async_to_lds_b128 pipeline; gx and weights pre-swizzled into
// B-fragment layout (contiguous 32B/lane -> 2x b128 per fragment).
// ---------------------------------------------------------------------------

#define D_IN  64
#define D_HID 128
#define NB    8
#define MM    2048
#define ROWS  (NB * MM)   // 16384

typedef __attribute__((ext_vector_type(16))) __bf16 v16bf;
typedef __attribute__((ext_vector_type(8)))  float  v8f;

// --- CDNA5 async global->LDS copy (CDNA5 ISA) ------------
// LDS offset = low 32 bits of flat shared address (ISA 10.2).
__device__ __forceinline__ void async_load_b128(unsigned int lds_off,
                                                const void* gptr) {
  asm volatile("global_load_async_to_lds_b128 %0, %1, off th:TH_LOAD_NT"
               :: "v"(lds_off), "v"((unsigned long long)(uintptr_t)gptr)
               : "memory");
}
template <int N>
__device__ __forceinline__ void wait_async() {
  asm volatile("s_wait_asynccnt %0" :: "i"(N) : "memory");
}

// --- B-fragment swizzle -----------------------------------------------------
// B 32x16 bf16 fragment (contiguous k order used here, not ISA table order): lane l holds N = l&15, Kgroup g = l>=16,
// element e -> K = 16g+e.  Swizzled storage: frag F = kc*(N/16)+nt, within frag
// lane*16 halfs (32B contiguous per lane -> v16bf load = 2x b128).
__device__ __forceinline__ int swz_src(int d, int NOUT) {
  int F = d >> 9, r = d & 511, lane = r >> 4, e = r & 15;
  int ntiles = NOUT >> 4;
  int kc = F / ntiles, nt = F - kc * ntiles;
  int k = kc * 32 + ((lane >> 4) << 4) + e;
  int n = nt * 16 + (lane & 15);
  return k * NOUT + n;
}

// A-matrix 16x32 bf16 fragment from row-major f32 (16 rows, ld floats).
// lane L: row = L&15, g = L>=16; element e -> K = 16g+e (matches B swizzle).
__device__ __forceinline__ v16bf a_frag_from_f32(const float* __restrict__ tile,
                                                 int ld, int lane, int kbase) {
  const int g = (lane >> 4) & 1;
  const float* p = tile + (size_t)(lane & 15) * ld;
  v16bf a;
#pragma unroll
  for (int e = 0; e < 16; ++e) {
    int k = kbase + g * 16 + e;
    a[e] = (__bf16)p[k];
  }
  return a;
}

// --- weight prep: f32 -> bf16, B-frag swizzle, fold g_w1 halves -------------
__global__ void prep_weights_kernel(const float* __restrict__ fw1, const float* __restrict__ fw2,
                                    const float* __restrict__ gw1, const float* __restrict__ gw2,
                                    __bf16* __restrict__ o_fw1, __bf16* __restrict__ o_fw2,
                                    __bf16* __restrict__ o_gw1e, __bf16* __restrict__ o_gw2) {
  int d = blockIdx.x * blockDim.x + threadIdx.x;
  if (d < D_IN * D_IN)   o_fw1[d]  = (__bf16)fw1[swz_src(d, D_IN)];
  if (d < D_IN * D_HID) {
    int s = swz_src(d, D_HID);
    o_fw2[d]  = (__bf16)fw2[s];
    o_gw1e[d] = (__bf16)(gw1[s] + gw1[D_IN * D_HID + s]);  // cat(x,x) fold
  }
  if (d < D_HID * D_HID) o_gw2[d] = (__bf16)gw2[swz_src(d, D_HID)];
}

// --- generic MLP layer: Out = act(A[rows x K] @ W[K x NOUT] + bias) ---------
// One wave per 16-row tile; weights read as swizzled fragments from global.
template <int K, int NOUT, bool RELU, bool OUT_SWZ>
__global__ __launch_bounds__(128)
void mlp_layer_kernel(const float* __restrict__ A, const __bf16* __restrict__ Wswz,
                      const float* __restrict__ bias, void* __restrict__ Out, int rows) {
  const int tid = threadIdx.x;
  const int wave = tid >> 5, lane = tid & 31;
  const int row0 = (blockIdx.x * 4 + wave) * 16;
  if (row0 >= rows) return;

  v16bf afr[K / 32];
#pragma unroll
  for (int kc = 0; kc < K / 32; ++kc)
    afr[kc] = a_frag_from_f32(A + (size_t)row0 * K, K, lane, kc * 32);

  const int g = lane >> 4, nl = lane & 15;
#pragma unroll
  for (int nt = 0; nt < NOUT / 16; ++nt) {
    v8f acc = {};
    v16bf b[K / 32];
#pragma unroll
    for (int kc = 0; kc < K / 32; ++kc)
      b[kc] = *(const v16bf*)(Wswz + ((size_t)(kc * (NOUT / 16) + nt) << 9) + lane * 16);
    __builtin_amdgcn_sched_barrier(0);  // keep all loads in flight before WMMAs
#pragma unroll
    for (int kc = 0; kc < K / 32; ++kc)
      acc = __builtin_amdgcn_wmma_f32_16x16x32_bf16(false, afr[kc], false, b[kc],
                                                    (short)0, acc, false, false);
    const int n = nt * 16 + nl;
    const float bv = bias[n];
#pragma unroll
    for (int r = 0; r < 8; ++r) {
      float v = acc[r] + bv;
      if (RELU) v = fmaxf(v, 0.0f);
      const int row = row0 + g * 8 + r;
      if (OUT_SWZ) {
        // scatter into einsum B-fragment layout: i = row%MM, h = n
        const int nb = row >> 11, ib = row & (MM - 1);
        const int it = ib >> 5, gp = (ib >> 4) & 1, e = ib & 15;
        const size_t d = ((((size_t)nb * (MM / 32) + it) * (D_HID / 16) + (n >> 4)) << 9)
                       + (((n & 15) + 16 * gp) << 4) + e;
        ((__bf16*)Out)[d] = (__bf16)v;
      } else {
        ((float*)Out)[(size_t)row * NOUT + n] = v;
      }
    }
  }
}

// --- einsum: out[n,j,h] = sd[n,j,h] + sum_i edge[n,i,j] * gx[n,i,h] ---------
// 4 independent waves/block, no barriers. Edge tiles [32 x 16] f32 stream
// through a 4-buffer LDS ring via async-to-LDS, issued 3 chunks ahead
// (s_wait_asynccnt 12 with in-order completion => oldest chunk landed).
// Per chunk: A = edge^T (LDS gather + cvt), 8 B frags as contiguous v16bf
// from swizzled gx (L2-hot); sched_barrier keeps all 16 B-loads in flight
// so WMMA t overlaps loads t+1..7 via incremental s_wait_loadcnt.
__global__ __launch_bounds__(128)
void einsum_kernel(const float* __restrict__ edge, const __bf16* __restrict__ gx_swz,
                   const float* __restrict__ sd, float* __restrict__ out) {
  __shared__ float edge_s[4][4][32 * 16];   // [wave][ring buf][tile] = 32 KB

  const int tid = threadIdx.x, wave = tid >> 5, lane = tid & 31;
  const int n = blockIdx.x >> 5;
  const int jbase = ((blockIdx.x & 31) * 4 + wave) * 16;

  const float*  E  = edge + (size_t)n * MM * MM;
  const __bf16* Gs = gx_swz + ((size_t)n * (MM / 32) * (D_HID / 16) << 9);

  const unsigned int lds_lane =
      (unsigned int)(uintptr_t)(&edge_s[wave][0][0]) + (unsigned int)lane * 64u;

  v8f acc[8];
#pragma unroll
  for (int t = 0; t < 8; ++t) acc[t] = (v8f){};

  const int g = lane >> 4, nl = lane & 15;

  auto issue = [&](int c) {
    const char* gp = (const char*)(E + (((size_t)c << 5) + lane) * MM + jbase);
    const unsigned int b = lds_lane + (unsigned int)(c & 3) * 2048u;
#pragma unroll
    for (int q = 0; q < 4; ++q) async_load_b128(b + q * 16u, gp + q * 16);
    if (c + 8 < MM / 32)                    // warm L2 well beyond async distance
      __builtin_prefetch(gp + (size_t)8 * 32 * MM * 4, 0, 1);
  };

  auto compute = [&](int c) {
    const float* es = &edge_s[wave][c & 3][0];
    // A = edge^T tile: A[j,k] = es[k*16 + j], cvt f32->bf16
    v16bf a;
#pragma unroll
    for (int e = 0; e < 16; ++e) {
      int k = g * 16 + e;
      a[e] = (__bf16)es[k * 16 + nl];
    }
    const __bf16* bp = Gs + (((size_t)c * (D_HID / 16)) << 9) + lane * 16;
    v16bf b[8];
#pragma unroll
    for (int t = 0; t < 8; ++t) b[t] = *(const v16bf*)(bp + ((size_t)t << 9));
    __builtin_amdgcn_sched_barrier(0);  // all 16 loads issued before any WMMA
#pragma unroll
    for (int t = 0; t < 8; ++t)
      acc[t] = __builtin_amdgcn_wmma_f32_16x16x32_bf16(false, a, false, b[t],
                                                       (short)0, acc[t], false, false);
  };

  issue(0); issue(1); issue(2);
  for (int c = 0; c < MM / 32 - 3; ++c) {   // c = 0..60
    issue(c + 3);
    wait_async<12>();
    compute(c);
  }
  wait_async<8>();  compute(MM / 32 - 3);
  wait_async<4>();  compute(MM / 32 - 2);
  wait_async<0>();  compute(MM / 32 - 1);

  // out = acc + self_dyn  (D layout: M = r + 8g, N = lane&15)
#pragma unroll
  for (int t = 0; t < 8; ++t) {
    const int h = t * 16 + nl;
#pragma unroll
    for (int r = 0; r < 8; ++r) {
      const int j = jbase + g * 8 + r;
      const size_t idx = ((size_t)n * MM + j) * (size_t)D_HID + h;
      out[idx] = acc[t][r] + sd[idx];
    }
  }
}

// ---------------------------------------------------------------------------
extern "C" void kernel_launch(void* const* d_in, const int* in_sizes, int n_in,
                              void* d_out, int out_size, void* d_ws, size_t ws_size,
                              hipStream_t stream) {
  const float* x    = (const float*)d_in[0];
  const float* edge = (const float*)d_in[1];
  const float* f_w1 = (const float*)d_in[2];
  const float* f_b1 = (const float*)d_in[3];
  const float* f_w2 = (const float*)d_in[4];
  const float* f_b2 = (const float*)d_in[5];
  const float* g_w1 = (const float*)d_in[6];
  const float* g_b1 = (const float*)d_in[7];
  const float* g_w2 = (const float*)d_in[8];
  const float* g_b2 = (const float*)d_in[9];
  float* out = (float*)d_out;

  char* ws = (char*)d_ws;
  __bf16* w_fw1  = (__bf16*)(ws + 0);          // 64*64*2   =  8 KB (swizzled)
  __bf16* w_fw2  = (__bf16*)(ws + (8 << 10));  // 64*128*2  = 16 KB
  __bf16* w_gw1e = (__bf16*)(ws + (24 << 10)); // 64*128*2  = 16 KB
  __bf16* w_gw2  = (__bf16*)(ws + (40 << 10)); // 128*128*2 = 32 KB
  size_t off = 128 << 10;
  __bf16* gxbuf = (__bf16*)(ws + off); off += (size_t)ROWS * D_HID * 2; // 4 MB swz
  float*  sdbuf = (float* )(ws + off); off += (size_t)ROWS * D_HID * 4; // 8 MB
  float*  hid   = (float* )(ws + off);                                  // 8 MB

  // 0) weights -> bf16, B-fragment swizzle (fold g_w1 halves)
  prep_weights_kernel<<<64, 256, 0, stream>>>(f_w1, f_w2, g_w1, g_w2,
                                              w_fw1, w_fw2, w_gw1e, w_gw2);
  // 1) hid = relu(x @ fw1 + fb1)            [ROWS x 64]
  mlp_layer_kernel<64, 64, true, false><<<256, 128, 0, stream>>>(x, w_fw1, f_b1, hid, ROWS);
  // 2) sd = hid @ fw2 + fb2                 [ROWS x 128] f32
  mlp_layer_kernel<64, 128, false, false><<<256, 128, 0, stream>>>(hid, w_fw2, f_b2, sdbuf, ROWS);
  // 3) hid = relu(x @ gw1_eff + gb1)        [ROWS x 128]
  mlp_layer_kernel<64, 128, true, false><<<256, 128, 0, stream>>>(x, w_gw1e, g_b1, hid, ROWS);
  // 4) gx = hid @ gw2 + gb2 -> swizzled bf16 B-fragments
  mlp_layer_kernel<128, 128, false, true><<<256, 128, 0, stream>>>(hid, w_gw2, g_b2, gxbuf, ROWS);
  // 5) out = sd + edge^T @ gx
  einsum_kernel<<<NB * (MM / 64), 128, 0, stream>>>(edge, gxbuf, sdbuf, out);
}
